// SelfAttentionV1_26860725469570
// MI455X (gfx1250) — compile-verified, matrix operand order checked by reading, not run on hardware
//
#include <hip/hip_runtime.h>
#include <hip/hip_bf16.h>

// ---------------------------------------------------------------------------
// Self-attention (b=4, s=4096, d=768) for gfx1250 using v_wmma_f32_16x16x32_bf16
// Stage 0: fp32 -> bf16 conversion of x and weights
// Stage 1: Q = x Wq^T + bq, K = x Wk^T + bk  (row-major [s,d], K-contiguous)
//          Vt = Wv x^T + bv                  ([d, b*s], key-contiguous)
// Stage 2: flash attention, 64-query tile per workgroup, streaming softmax.
//          Q tile staged once in LDS (reused 64x); K/V streamed from global
//          through rolled ping-pong loops that force prefetch distance.
// ---------------------------------------------------------------------------

#define HID   768
#define SEQ   4096
#define NBAT  4
#define MROWS (NBAT * SEQ)          // 16384
#define SM_SCALE 0.03608439182435161f   // 1/sqrt(768)

typedef __attribute__((ext_vector_type(16))) __bf16 v16bf;
typedef __attribute__((ext_vector_type(8)))  __bf16 v8bf;
typedef __attribute__((ext_vector_type(8)))  float  v8f;

union V16U { v16bf v; v8bf h[2]; };

__device__ __forceinline__ v16bf mk16(v8bf lo, v8bf hi) {
    V16U u; u.h[0] = lo; u.h[1] = hi; return u.v;
}

__device__ __forceinline__ unsigned short f2bf(float f) {
    unsigned int u = __float_as_uint(f);
    unsigned int r = u + 0x7FFFu + ((u >> 16) & 1u);   // round-to-nearest-even
    return (unsigned short)(r >> 16);
}

// Per-lane fragment fetch: lane (l&15) = row of a K-contiguous matrix,
// chunks K=[koff..koff+7] and K=[koff+16..koff+23] with koff = (l>>4)*8.
// Matches the CDNA5 16-bit A-operand layout; B operand = same pattern on N x K.
__device__ __forceinline__ v16bf load_frag(const unsigned short* p) {
    return mk16(*(const v8bf*)p, *(const v8bf*)(p + 16));
}

#define WMMA_BF16(A, B, C) __builtin_amdgcn_wmma_f32_16x16x32_bf16( \
    false, (A), false, (B), (short)0, (C), false, false)

// ---------------------------------------------------------------------------
// Stage 0: fp32 -> bf16 (4-wide)
// ---------------------------------------------------------------------------
struct alignas(8) US4 { unsigned short x, y, z, w; };

__global__ void cvt_f32_bf16(const float* __restrict__ in,
                             unsigned short* __restrict__ out, int n4) {
    int i = blockIdx.x * blockDim.x + threadIdx.x;
    if (i < n4) {
        float4 f = ((const float4*)in)[i];
        US4 o; o.x = f2bf(f.x); o.y = f2bf(f.y); o.z = f2bf(f.z); o.w = f2bf(f.w);
        ((US4*)out)[i] = o;
    }
}

// ---------------------------------------------------------------------------
// Stage 1: C[M,N] = A[M,768] * B[N,768]^T + bias, bf16 in / f32 acc / bf16 out.
// Block = 128 threads (4 waves), tile 64(M) x 64(N); wave w -> rows w*16..+16.
// bias_on_m==0: bias[col] (Q/K projections); ==1: bias[row] (Vt).
// Rolled ping-pong k-loop: loop-carried buffers force one-iteration prefetch
// distance (staggered s_wait_loadcnt instead of wait-0 per WMMA).
// ---------------------------------------------------------------------------
__global__ __launch_bounds__(128)
void gemm_bf16(const unsigned short* __restrict__ A,
               const unsigned short* __restrict__ B,
               const float* __restrict__ bias,
               unsigned short* __restrict__ C,
               int ldc, int bias_on_m) {
    const int lane = threadIdx.x & 31;
    const int wave = threadIdx.x >> 5;
    const int mbase = blockIdx.y * 64 + wave * 16;
    const int nbase = blockIdx.x * 64;
    const int rlo  = lane & 15;
    const int koff = (lane >> 4) * 8;

    v8f acc[4] = {};

    const unsigned short* aptr = A + (size_t)(mbase + rlo) * HID + koff;
    const unsigned short* bptr = B + (size_t)(nbase + rlo) * HID + koff;

    // ping-pong buffers: set 0 holds k-step kk, set 1 holds kk+32
    v16bf a0 = load_frag(aptr);
    v16bf a1 = load_frag(aptr + 32);
    v16bf b0[4], b1[4];
#pragma unroll
    for (int nt = 0; nt < 4; ++nt) b0[nt] = load_frag(bptr + nt * 16 * HID);
#pragma unroll
    for (int nt = 0; nt < 4; ++nt) b1[nt] = load_frag(bptr + nt * 16 * HID + 32);

    for (int kk = 0; kk < HID - 64; kk += 64) {   // rolled on purpose
#pragma unroll
        for (int nt = 0; nt < 4; ++nt) acc[nt] = WMMA_BF16(a0, b0[nt], acc[nt]);
        a0 = load_frag(aptr + kk + 64);
#pragma unroll
        for (int nt = 0; nt < 4; ++nt)
            b0[nt] = load_frag(bptr + nt * 16 * HID + kk + 64);
#pragma unroll
        for (int nt = 0; nt < 4; ++nt) acc[nt] = WMMA_BF16(a1, b1[nt], acc[nt]);
        a1 = load_frag(aptr + kk + 96);
#pragma unroll
        for (int nt = 0; nt < 4; ++nt)
            b1[nt] = load_frag(bptr + nt * 16 * HID + kk + 96);
    }
#pragma unroll
    for (int nt = 0; nt < 4; ++nt) acc[nt] = WMMA_BF16(a0, b0[nt], acc[nt]);
#pragma unroll
    for (int nt = 0; nt < 4; ++nt) acc[nt] = WMMA_BF16(a1, b1[nt], acc[nt]);

    const int rhi = (lane >> 4) * 8;   // C/D layout: lanes 16-31 hold M+8
#pragma unroll
    for (int nt = 0; nt < 4; ++nt) {
        int col = nbase + nt * 16 + rlo;
#pragma unroll
        for (int r = 0; r < 8; ++r) {
            int row = mbase + r + rhi;
            float v = acc[nt][r] + (bias_on_m ? bias[row] : bias[col]);
            C[(size_t)row * ldc + col] = f2bf(v);
        }
    }
}

// ---------------------------------------------------------------------------
// Stage 2: flash attention. Block = 512 threads (16 waves), 64-query tile.
// Wave id w: mt = w&3 (16-row m-tile), gg = w>>2.
//   S phase : wave (mt,gg) computes S[mt*16.., gg*16..] over full 768 dims,
//             A-operand (Q) from LDS, B-operand (K) streamed from global.
//   PV phase: wave (mt,gg) owns d-slice [gg*192, +192) -> 12 accumulators.
// ---------------------------------------------------------------------------
#define S_LD 68
#define P_LD 72
#define Q_LD 776    // 768 + 8 pad; row stride bytes = 1552 (16B aligned)

__global__ __launch_bounds__(512)
void attn_kernel(const unsigned short* __restrict__ Q,
                 const unsigned short* __restrict__ Km,
                 const unsigned short* __restrict__ Vt,
                 float* __restrict__ out) {
    __shared__ unsigned short Qs[64 * Q_LD];   // 97 KB: Q tile, reused 64x
    __shared__ float          Sl[64 * S_LD];
    __shared__ unsigned short Pl[64 * P_LD];
    __shared__ float rowm[64], rowl[64], ralpha[64];
    __shared__ float pmax[64 * 8], psum[64 * 8];

    const int tid  = threadIdx.x;
    const int lane = tid & 31;
    const int wave = tid >> 5;
    const int mt = wave & 3;
    const int gg = wave >> 2;
    const int b  = blockIdx.y;
    const int q0 = blockIdx.x * 64;

    const int rlo  = lane & 15;
    const int koff = (lane >> 4) * 8;
    const int rhi  = (lane >> 4) * 8;

    const int srow = tid >> 3;   // softmax: one row per 8 threads
    const int sseg = tid & 7;

    if (tid < 64) { rowm[tid] = -1e30f; rowl[tid] = 0.0f; }

    // ---- stage Q tile into LDS once (coalesced b128 copy) ----
    {
        const unsigned short* qg = Q + (size_t)(b * SEQ + q0) * HID;
        for (int i = tid; i < 64 * (HID / 8); i += 512) {   // 6144 chunks of 8
            int row = i / (HID / 8);
            int col = (i % (HID / 8)) * 8;
            *(v8bf*)&Qs[row * Q_LD + col] = *(const v8bf*)(qg + (size_t)row * HID + col);
        }
    }

    v8f o[12] = {};

    const unsigned short* qlds = &Qs[(mt * 16 + rlo) * Q_LD + koff];

    __syncthreads();   // Q staged + stats initialized

    for (int kv0 = 0; kv0 < SEQ; kv0 += 64) {
        const unsigned short* krow =
            Km + (size_t)(b * SEQ + kv0 + gg * 16 + rlo) * HID + koff;

        // warm WGP$/L2 one key-tile ahead (global_prefetch_b8)
        if (kv0 + 64 < SEQ) {
            __builtin_prefetch(krow + (size_t)64 * HID, 0, 3);
            __builtin_prefetch(Vt + (size_t)(gg * 192 + rlo) * MROWS
                                  + (b * SEQ + kv0 + 64) + koff, 0, 3);
        }

        // ---- S tile: [16 q] x [16 keys]; Q from LDS, K ping-pong from global ----
        v8f s = {};
        v16bf kb0 = load_frag(krow);
        v16bf kb1 = load_frag(krow + 32);
        for (int kk = 0; kk < HID - 64; kk += 64) {   // rolled on purpose
            v16bf qa0 = load_frag(qlds + kk);
            s = WMMA_BF16(qa0, kb0, s);
            kb0 = load_frag(krow + kk + 64);
            v16bf qa1 = load_frag(qlds + kk + 32);
            s = WMMA_BF16(qa1, kb1, s);
            kb1 = load_frag(krow + kk + 96);
        }
        {
            v16bf qa0 = load_frag(qlds + (HID - 64));
            s = WMMA_BF16(qa0, kb0, s);
            v16bf qa1 = load_frag(qlds + (HID - 32));
            s = WMMA_BF16(qa1, kb1, s);
        }

#pragma unroll
        for (int r = 0; r < 8; ++r)
            Sl[(mt * 16 + r + rhi) * S_LD + gg * 16 + rlo] = s[r];
        __syncthreads();                                   // (1) S complete

        // ---- streaming softmax: 8 threads per row ----
        float sc[8];
        float mloc = -1e30f;
#pragma unroll
        for (int c = 0; c < 8; ++c) {
            float v = Sl[srow * S_LD + sseg * 8 + c] * SM_SCALE;
            sc[c] = v; mloc = fmaxf(mloc, v);
        }
        pmax[srow * 8 + sseg] = mloc;
        __syncthreads();                                   // (2) partial maxes
        if (sseg == 0) {
            float t = pmax[srow * 8];
#pragma unroll
            for (int c = 1; c < 8; ++c) t = fmaxf(t, pmax[srow * 8 + c]);
            float mprev = rowm[srow];
            float mnew  = fmaxf(mprev, t);
            rowm[srow]   = mnew;
            ralpha[srow] = __expf(mprev - mnew);
        }
        __syncthreads();                                   // (3) m/alpha ready
        {
            float mnew = rowm[srow];
            float ssum = 0.0f;
#pragma unroll
            for (int c = 0; c < 8; ++c) {
                float e = __expf(sc[c] - mnew);
                ssum += e;
                Pl[srow * P_LD + sseg * 8 + c] = f2bf(e);
            }
            psum[srow * 8 + sseg] = ssum;
        }
        __syncthreads();                                   // (4) P + partial sums
        if (sseg == 0) {
            float t = 0.0f;
#pragma unroll
            for (int c = 0; c < 8; ++c) t += psum[srow * 8 + c];
            rowl[srow] = ralpha[srow] * rowl[srow] + t;     // same thread updates
        }

        // ---- O = alpha*O + P @ V, V fragments ping-ponged across d-tiles ----
        float al[8];
#pragma unroll
        for (int r = 0; r < 8; ++r) al[r] = ralpha[mt * 16 + r + rhi];

        const unsigned short* prow = &Pl[(mt * 16 + rlo) * P_LD + koff];
        v16bf pa0 = load_frag(prow);        // keys 0..31
        v16bf pa1 = load_frag(prow + 32);   // keys 32..63

        const unsigned short* vbase =
            Vt + (size_t)(gg * 192 + rlo) * MROWS + (b * SEQ + kv0) + koff;

        v16bf v00 = load_frag(vbase);                           // t = 0
        v16bf v01 = load_frag(vbase + 32);
        v16bf v10 = load_frag(vbase + (size_t)16 * MROWS);      // t = 1
        v16bf v11 = load_frag(vbase + (size_t)16 * MROWS + 32);

#pragma unroll
        for (int t = 0; t < 12; t += 2) {
            {
                v8f oo = o[t];
#pragma unroll
                for (int r = 0; r < 8; ++r) oo[r] *= al[r];
                oo = WMMA_BF16(pa0, v00, oo);
                oo = WMMA_BF16(pa1, v01, oo);
                o[t] = oo;
            }
            if (t + 2 < 12) {
                const unsigned short* vp = vbase + (size_t)(t + 2) * 16 * MROWS;
                v00 = load_frag(vp);
                v01 = load_frag(vp + 32);
            }
            {
                v8f oo = o[t + 1];
#pragma unroll
                for (int r = 0; r < 8; ++r) oo[r] *= al[r];
                oo = WMMA_BF16(pa0, v10, oo);
                oo = WMMA_BF16(pa1, v11, oo);
                o[t + 1] = oo;
            }
            if (t + 3 < 12) {
                const unsigned short* vp = vbase + (size_t)(t + 3) * 16 * MROWS;
                v10 = load_frag(vp);
                v11 = load_frag(vp + 32);
            }
        }
        // next iteration's barriers protect S/P/alpha reuse; no extra sync needed
    }

    __syncthreads();   // make final rowl visible to all waves

    float linv[8];
#pragma unroll
    for (int r = 0; r < 8; ++r) linv[r] = 1.0f / rowl[mt * 16 + r + rhi];

#pragma unroll
    for (int t = 0; t < 12; ++t) {
        int col = gg * 192 + t * 16 + rlo;
#pragma unroll
        for (int r = 0; r < 8; ++r) {
            int row = mt * 16 + r + rhi;
            out[(size_t)(b * SEQ + q0 + row) * HID + col] = o[t][r] * linv[r];
        }
    }
}

// ---------------------------------------------------------------------------
// Launcher. Workspace layout (bf16 = unsigned short), total ~104 MB:
//   xb[16384*768] wqb[768*768] wkb wvb qb[16384*768] kb vtb[768*16384]
// ---------------------------------------------------------------------------
extern "C" void kernel_launch(void* const* d_in, const int* in_sizes, int n_in,
                              void* d_out, int out_size, void* d_ws, size_t ws_size,
                              hipStream_t stream) {
    const float* x  = (const float*)d_in[0];
    const float* Wq = (const float*)d_in[1];
    const float* bq = (const float*)d_in[2];
    const float* Wk = (const float*)d_in[3];
    const float* bk = (const float*)d_in[4];
    const float* Wv = (const float*)d_in[5];
    const float* bv = (const float*)d_in[6];
    float* out = (float*)d_out;

    const size_t NX = (size_t)MROWS * HID;   // 12,582,912
    const size_t NW = (size_t)HID * HID;     //    589,824

    unsigned short* xb  = (unsigned short*)d_ws;
    unsigned short* wqb = xb  + NX;
    unsigned short* wkb = wqb + NW;
    unsigned short* wvb = wkb + NW;
    unsigned short* qb  = wvb + NW;
    unsigned short* kb  = qb  + NX;
    unsigned short* vtb = kb  + NX;

    // Stage 0: conversions
    {
        int n4 = (int)(NX / 4);
        cvt_f32_bf16<<<(n4 + 255) / 256, 256, 0, stream>>>(x, xb, n4);
        int w4 = (int)(NW / 4);
        cvt_f32_bf16<<<(w4 + 255) / 256, 256, 0, stream>>>(Wq, wqb, w4);
        cvt_f32_bf16<<<(w4 + 255) / 256, 256, 0, stream>>>(Wk, wkb, w4);
        cvt_f32_bf16<<<(w4 + 255) / 256, 256, 0, stream>>>(Wv, wvb, w4);
    }

    // Stage 1: projections
    {
        dim3 gqk(HID / 64, MROWS / 64);      // (12, 256)
        gemm_bf16<<<gqk, 128, 0, stream>>>(xb, wqb, bq, qb, HID, 0);
        gemm_bf16<<<gqk, 128, 0, stream>>>(xb, wkb, bk, kb, HID, 0);
        dim3 gv(MROWS / 64, HID / 64);       // (256, 12)  : Vt = Wv * x^T
        gemm_bf16<<<gv, 128, 0, stream>>>(wvb, xb, bv, vtb, MROWS, 1);
    }

    // Stage 2: attention
    {
        dim3 ga(SEQ / 64, NBAT);             // (64, 4)
        attn_kernel<<<ga, 512, 0, stream>>>(qb, kb, vtb, out);
    }
}
